// GraphAttention_12859132084636
// MI455X (gfx1250) — matrix operand order repaired and checked by reading, not compiled, not run on hardware
//
#include <hip/hip_runtime.h>

#define B_SZ  4096
#define N_SZ  4096
#define NINP  64
#define NFEAT 32
#define NH    4

typedef _Float16 half16 __attribute__((ext_vector_type(16)));
typedef _Float16 half2t __attribute__((ext_vector_type(2)));
typedef float    float8 __attribute__((ext_vector_type(8)));

// ---------------------------------------------------------------------------
// Kernel 1: projection. WG[h,f,n] (f16, column-major per (h,f) so kernel 2's
// WMMA B-fragment K-pairs are contiguous) and s_dst[h,n] = WG . a_dst.
// ---------------------------------------------------------------------------
__global__ __launch_bounds__(256) void gat_proj_kernel(
    const float* __restrict__ G, const float* __restrict__ W,
    const float* __restrict__ bias, const float* __restrict__ a_dst,
    _Float16* __restrict__ WGcm, float* __restrict__ sdst)
{
    __shared__ float Wl[NH * NINP * NFEAT];   // 32 KB
    __shared__ float bl[NH * NFEAT];
    __shared__ float al[NH * NFEAT];
    const int tid = threadIdx.x;
    for (int i = tid; i < NH * NINP * NFEAT; i += 256) Wl[i] = W[i];
    for (int i = tid; i < NH * NFEAT; i += 256) { bl[i] = bias[i]; al[i] = a_dst[i]; }
    __syncthreads();

    const int n = blockIdx.x * 256 + tid;
    float g[NINP];
#pragma unroll
    for (int i = 0; i < NINP; ++i) g[i] = G[(size_t)n * NINP + i];

    for (int h = 0; h < NH; ++h) {
        float sh = 0.0f;
        for (int f = 0; f < NFEAT; ++f) {
            float acc = bl[h * NFEAT + f];
#pragma unroll
            for (int i = 0; i < NINP; ++i)
                acc += g[i] * Wl[h * NINP * NFEAT + i * NFEAT + f];
            sh += acc * al[h * NFEAT + f];
            WGcm[(size_t)(h * NFEAT + f) * N_SZ + n] = (_Float16)acc;
        }
        sdst[(size_t)h * N_SZ + n] = sh;
    }
}

// ---------------------------------------------------------------------------
// Kernel 2: fused masked-softmax + aggregation, split-K over the softmax axis.
// Workgroup = one 16-row x 32-col output tile; 4 waves each cover N/4,
// partial C-fragments and row-sums Z reduced through LDS.
// ---------------------------------------------------------------------------
__global__ __launch_bounds__(128) void gat_attn_kernel(
    const float* __restrict__ A, const float* __restrict__ sdst,
    const _Float16* __restrict__ WGcm, float* __restrict__ out)
{
    __shared__ float sd[N_SZ];                // 16 KB: s_dst for this head
    __shared__ float cbuf[4][2][8][32];       // 8 KB: per-wave C partials
    __shared__ float zbuf[4][16];             // per-wave row-sum partials

    const int h   = blockIdx.y;
    const int tid = threadIdx.x;

    // Stage s_dst[h, :] into LDS via the gfx1250 async-to-LDS path (ASYNCcnt).
    // Toolchain signature (from diagnostics): (addrspace(1) v4i*, lds v4i*, imm, imm)
#if __has_builtin(__builtin_amdgcn_global_load_async_to_lds_b128)
    {
        typedef int v4i __attribute__((vector_size(16)));
        typedef __attribute__((address_space(1))) v4i gv4i;
        typedef __attribute__((address_space(3))) v4i lv4i;
        const float* src = sdst + (size_t)h * N_SZ;
#pragma unroll
        for (int t = 0; t < N_SZ / 4; t += 128) {
            const int idx = (t + tid) * 4;
            __builtin_amdgcn_global_load_async_to_lds_b128(
                (gv4i*)(src + idx), (lv4i*)(&sd[idx]), 0, 0);
        }
#if __has_builtin(__builtin_amdgcn_s_wait_asynccnt)
        __builtin_amdgcn_s_wait_asynccnt(0);
#endif
    }
#else
    for (int t = tid; t < N_SZ; t += 128) sd[t] = sdst[(size_t)h * N_SZ + t];
#endif
    __syncthreads();

    const int lane = tid & 31;
    const int wave = tid >> 5;
    const int hi   = lane >> 4;               // lane half (K-group select)
    const int mrow = lane & 15;               // A-frag row / B-frag col
    const int brow = blockIdx.x * 16 + mrow;
    const float* __restrict__ Arow = A + (size_t)brow * N_SZ;

    float8 c0 = {};
    float8 c1 = {};
    float  z  = 0.0f;

    // Split-K: wave w covers n = w*32, stride 128.
    for (int nb = wave * 32; nb < N_SZ; nb += 128) {
        __builtin_prefetch(Arow + nb + 128, 0, 1);   // global_prefetch_b8

        // --- A fragment: p = exp(sdst - A), 16x32 f16, k = 16*hi + s (contiguous) ---
        half16 afrag;
#pragma unroll
        for (int v = 0; v < 8; ++v) {
            const int k = (hi << 4) + (v << 1);
            const int n = nb + k;
            const float2 av = *reinterpret_cast<const float2*>(Arow + n);
            const float2 sv = *reinterpret_cast<const float2*>(&sd[n]);
            const float p0 = __expf(sv.x - av.x);
            const float p1 = __expf(sv.y - av.y);
            z += p0 + p1;
            afrag[2 * v]     = (_Float16)p0;
            afrag[2 * v + 1] = (_Float16)p1;
        }

        // --- B fragments: WG columns (f, f+16), K-pair contiguous in memory ---
        half16 b0, b1;
#pragma unroll
        for (int v = 0; v < 8; ++v) {
            const int n = nb + (v << 1) + (hi << 4);
            const half2t p0 = *reinterpret_cast<const half2t*>(
                WGcm + (size_t)(h * NFEAT + mrow) * N_SZ + n);
            const half2t p1 = *reinterpret_cast<const half2t*>(
                WGcm + (size_t)(h * NFEAT + 16 + mrow) * N_SZ + n);
            b0[2 * v] = p0.x; b0[2 * v + 1] = p0.y;
            b1[2 * v] = p1.x; b1[2 * v + 1] = p1.y;
        }

        c0 = __builtin_amdgcn_wmma_f32_16x16x32_f16(false, afrag, false, b0,
                                                    (short)0, c0, false, false);
        c1 = __builtin_amdgcn_wmma_f32_16x16x32_f16(false, afrag, false, b1,
                                                    (short)0, c1, false, false);
    }

    // Lanes r and r+16 hold complementary K-halves of row r's Z -> combine.
    z += __builtin_bit_cast(float,
            __builtin_amdgcn_ds_bpermute((lane ^ 16) << 2,
                                         __builtin_bit_cast(int, z)));

    // Deposit per-wave partials for the split-K reduction.
#pragma unroll
    for (int j = 0; j < 8; ++j) {
        cbuf[wave][0][j][lane] = c0[j];
        cbuf[wave][1][j][lane] = c1[j];
    }
    if (lane < 16) zbuf[wave][lane] = z;
    __syncthreads();

    if (wave == 0) {
        const int bbase = blockIdx.x * 16;
#pragma unroll
        for (int j = 0; j < 8; ++j) {
            const int row = j + (hi << 3);
            const float zr = zbuf[0][row] + zbuf[1][row] +
                             zbuf[2][row] + zbuf[3][row];
#if __has_builtin(__builtin_amdgcn_rcpf)
            const float inv = __builtin_amdgcn_rcpf(zr);
#else
            const float inv = 1.0f / zr;
#endif
            const float s0 = cbuf[0][0][j][lane] + cbuf[1][0][j][lane] +
                             cbuf[2][0][j][lane] + cbuf[3][0][j][lane];
            const float s1 = cbuf[0][1][j][lane] + cbuf[1][1][j][lane] +
                             cbuf[2][1][j][lane] + cbuf[3][1][j][lane];
            float o0 = s0 * inv;  o0 = o0 > 0.0f ? o0 : 0.0f;
            float o1 = s1 * inv;  o1 = o1 > 0.0f ? o1 : 0.0f;
            const size_t base =
                (size_t)(bbase + row) * (NH * NFEAT) + h * NFEAT + mrow;
            out[base]      = o0;
            out[base + 16] = o1;
        }
    }
}

// ---------------------------------------------------------------------------
extern "C" void kernel_launch(void* const* d_in, const int* in_sizes, int n_in,
                              void* d_out, int out_size, void* d_ws, size_t ws_size,
                              hipStream_t stream) {
    // setup_inputs order: X(0), G(1), A(2), W(3), b(4), a_src(5), a_dst(6), att_b(7)
    // (X, a_src, att_b drop out of softmax -> unused.)
    const float* G     = (const float*)d_in[1];
    const float* A     = (const float*)d_in[2];
    const float* W     = (const float*)d_in[3];
    const float* bias  = (const float*)d_in[4];
    const float* a_dst = (const float*)d_in[6];
    float*       out   = (float*)d_out;

    _Float16* WGcm = (_Float16*)d_ws;                                        // 1 MB
    float*    sdst = (float*)((char*)d_ws +
                              (size_t)NH * NFEAT * N_SZ * sizeof(_Float16)); // 64 KB

    gat_proj_kernel<<<dim3(N_SZ / 256), dim3(256), 0, stream>>>(
        G, W, bias, a_dst, WGcm, sdst);
    gat_attn_kernel<<<dim3(B_SZ / 16, NH), dim3(128), 0, stream>>>(
        A, sdst, WGcm, out);
}
